// AttFKANBlock_10290741641645
// MI455X (gfx1250) — matrix operand order repaired and checked by reading, not compiled, not run on hardware
//
#include <hip/hip_runtime.h>
#include <hip/hip_bf16.h>

typedef __attribute__((ext_vector_type(16))) _Float16 v16h;
typedef __attribute__((ext_vector_type(8)))  float    v8f;
typedef __attribute__((ext_vector_type(4)))  int      v4i;

typedef __attribute__((address_space(1))) v4i* gv4i_ptr;  // global
typedef __attribute__((address_space(3))) v4i* lv4i_ptr;  // LDS

#define TOKENS_TOTAL (16 * 4096)       // B*L = 65536
#define TILES        (TOKENS_TOTAL/16) // 4096 wave-tiles
#define ROWSTRIDE    132               // padded LDS row stride (bank-conflict free)
#define KSTEPS       64                // K=2048 / 32
#define LAYER_HALFS  (64*8*32*16)      // 262144 f16 per layer of packed B
#define KSTEP_HALFS  4096              // 8 nt * 32 lanes * 16 halfs per k-step

#if __has_builtin(__builtin_amdgcn_global_load_async_to_lds_b128)
#define USE_ASYNC_LDS 1
#else
#define USE_ASYNC_LDS 0
#endif

__device__ __forceinline__ void wait_async_done() {
#if USE_ASYNC_LDS
#if __has_builtin(__builtin_amdgcn_s_wait_asynccnt)
  __builtin_amdgcn_s_wait_asynccnt(0);
#else
  asm volatile("s_wait_asynccnt 0x0" ::: "memory");
#endif
#endif
}

// Stage one k-step's 8 KB of packed-B fragments global -> LDS (128 threads).
__device__ __forceinline__
void stage_b(const _Float16* __restrict__ gsrc, _Float16* lbuf, int tid) {
  #pragma unroll
  for (int r = 0; r < 4; ++r) {
    int q = tid + 128 * r;                 // 16-byte chunk id, 0..511
#if USE_ASYNC_LDS
    __builtin_amdgcn_global_load_async_to_lds_b128(
        (gv4i_ptr)((const char*)gsrc + q * 16),
        (lv4i_ptr)((char*)lbuf + q * 16),
        0, 0);
#else
    ((uint4*)lbuf)[q] = ((const uint4*)gsrc)[q];
#endif
  }
}

// ---------------------------------------------------------------------------
// Kernel 1: repack fk coeffs (2,O=128,I=128,G=8) f32 -> f16 WMMA-B fragments.
// Packed index P = ((((layer*64+ks)*8+nt)*32+lane)*16 + e),
// lane = column n%16 (+K-half select), Klocal = 16*(lane/16) + e (contiguous k order, same as A).
// ---------------------------------------------------------------------------
__global__ __launch_bounds__(256)
void pack_coeffs_kernel(const float* __restrict__ fk1,
                        const float* __restrict__ fk2,
                        _Float16* __restrict__ Bpack) {
  int P = blockIdx.x * 256 + threadIdx.x;          // 0 .. 524287
  int e  = P & 15;
  int ln = (P >> 4) & 31;
  int nt = (P >> 9) & 7;
  int ks = (P >> 12) & 63;
  int a  = P >> 18;
  int K  = ks * 32 + e + ((ln >> 4) << 4);         // global contraction index
  int t  = K >> 10;                                 // 0=cos coeffs, 1=sin coeffs
  int i  = (K >> 3) & 127;                          // input dim
  int g  = K & 7;                                   // harmonic
  int n  = nt * 16 + (ln & 15);                     // output dim
  const float* src = a ? fk2 : fk1;                 // (2,128,128,8)
  Bpack[P] = (_Float16)src[((t * 128 + n) * 128 + i) * 8 + g];
}

// ---------------------------------------------------------------------------
// LayerNorm of a 16x128 tile held in LDS (row stride ROWSTRIDE).
// ---------------------------------------------------------------------------
__device__ __forceinline__
void ln_tile(float* tb, int lane, const float* __restrict__ g,
             const float* __restrict__ b) {
  int m = lane & 15, half = lane >> 4;
  float* row = tb + m * ROWSTRIDE + half * 64;
  float s = 0.f, ss = 0.f;
  #pragma unroll 4
  for (int c = 0; c < 64; ++c) { float v = row[c]; s += v; ss += v * v; }
  s  += __shfl_xor(s, 16, 32);
  ss += __shfl_xor(ss, 16, 32);
  float mean = s * (1.f / 128.f);
  float var  = ss * (1.f / 128.f) - mean * mean;
  float rstd = rsqrtf(var + 1e-5f);
  #pragma unroll 4
  for (int c = 0; c < 64; ++c) {
    int col = half * 64 + c;
    row[c] = (row[c] - mean) * rstd * g[col] + b[col];
  }
}

// ---------------------------------------------------------------------------
// Kernel 2: fused LN1 -> FKAN1(WMMA) -> ReLU -> LN2 -> FKAN2(WMMA) -> +bias.
// 4 waves/block, one 16-token tile per wave. Packed-B fragments are staged
// into double-buffered LDS once per k-step and shared by all 4 waves
// (4x less global/L2 B traffic), via async global->LDS copies when available.
// ---------------------------------------------------------------------------
__global__ __launch_bounds__(128)
void fkan_main_kernel(const float* __restrict__ x,
                      const float* __restrict__ n1g, const float* __restrict__ n1b,
                      const float* __restrict__ fk1b,
                      const float* __restrict__ n2g, const float* __restrict__ n2b,
                      const float* __restrict__ fk2b,
                      const _Float16* __restrict__ Bpack,
                      float* __restrict__ F) {
  __shared__ float tiles[4 * 16 * ROWSTRIDE];                 // 33 KB
  __shared__ __align__(32) _Float16 bstage[2][KSTEP_HALFS];   // 16 KB
  const int tid  = threadIdx.x;
  const int wave = tid >> 5;
  const int lane = tid & 31;
  const int m    = lane & 15;
  const int half = lane >> 4;
  float* tb = tiles + wave * (16 * ROWSTRIDE);
  const int tile = blockIdx.x * 4 + wave;          // 0 .. 4095

  // cooperative coalesced load of the 16x128 input tile
  {
    const float* src = x + (size_t)tile * 2048;
    for (int e = lane; e < 2048; e += 32)
      tb[(e >> 7) * ROWSTRIDE + (e & 127)] = src[e];
  }
  __syncthreads();
  ln_tile(tb, lane, n1g, n1b);
  __syncthreads();

  v8f acc[8];
  for (int layer = 0; layer < 2; ++layer) {
    #pragma unroll
    for (int t = 0; t < 8; ++t)
      acc[t] = (v8f){0.f, 0.f, 0.f, 0.f, 0.f, 0.f, 0.f, 0.f};

    const _Float16* lb = Bpack + (size_t)layer * LAYER_HALFS;

    // prologue: stage k-step 0
    stage_b(lb, bstage[0], tid);
    wait_async_done();
    __syncthreads();

    for (int ks = 0; ks < KSTEPS; ++ks) {
      const _Float16* cur = bstage[ks & 1] + lane * 16;
      if (ks + 1 < KSTEPS)
        stage_b(lb + (size_t)(ks + 1) * KSTEP_HALFS, bstage[(ks + 1) & 1], tid);

      // Build the A fragment (contiguous k order: k = 16*half + 8*u + h) in registers.
      // Each lane covers two full input dims (all 8 harmonics each).
      v16h a;
      #pragma unroll
      for (int u = 0; u < 2; ++u) {
        int i8 = ks * 4 + 2 * half + u;            // K/8 block index (0..255)
        float xi = tb[m * ROWSTRIDE + (i8 & 127)];
        int strig = i8 >> 7;                       // 0 -> cos half, 1 -> sin half
        float s1, c1;
        __sincosf(xi, &s1, &c1);
        float ck = c1, sk = s1;                    // cos(k*x), sin(k*x), k=1..8
        #pragma unroll
        for (int h = 0; h < 8; ++h) {
          a[8 * u + h] = (_Float16)(strig ? sk : ck);
          float cn = ck * c1 - sk * s1;            // angle-sum recurrence
          float sn = sk * c1 + ck * s1;
          ck = cn; sk = sn;
        }
      }
      #pragma unroll
      for (int nt = 0; nt < 8; ++nt) {
        v16h bfrag = *(const v16h*)(cur + nt * 512);
        acc[nt] = __builtin_amdgcn_wmma_f32_16x16x32_f16(
            false, a, false, bfrag, (short)0, acc[nt], false, false);
      }
      wait_async_done();
      __syncthreads();                             // next buffer ready, cur consumed
    }

    const float* bias = layer ? fk2b : fk1b;
    #pragma unroll
    for (int nt = 0; nt < 8; ++nt) {
      #pragma unroll
      for (int r = 0; r < 8; ++r) {
        int row = r + half * 8;                    // ISA C/D layout
        int col = nt * 16 + (lane & 15);
        float v = acc[nt][r] + bias[col];
        if (layer == 0) v = fmaxf(v, 0.f);         // ReLU between layers
        tb[row * ROWSTRIDE + col] = v;
      }
    }
    __syncthreads();

    if (layer == 0) {
      ln_tile(tb, lane, n2g, n2b);
      __syncthreads();
    } else {
      float* dst = F + (size_t)tile * 2048;
      for (int e = lane; e < 2048; e += 32)
        dst[e] = tb[(e >> 7) * ROWSTRIDE + (e & 127)];
    }
  }
}

// ---------------------------------------------------------------------------
// Kernel 3: channel pooling. Raw reshape => channel c4 = contiguous chunk of
// 4096 floats. One block per (b,c4) => avg + max.
// ---------------------------------------------------------------------------
__global__ __launch_bounds__(256)
void pool_kernel(const float* __restrict__ F,
                 float* __restrict__ avgp, float* __restrict__ mxp) {
  __shared__ float ssum[256], smax[256];
  int bc = blockIdx.x;                             // b*128 + c4
  int t  = threadIdx.x;
  const float* src = F + (size_t)bc * 4096;
  float s = 0.f, mx = -3.402823466e38f;
  for (int i = t; i < 4096; i += 256) { float v = src[i]; s += v; mx = fmaxf(mx, v); }
  ssum[t] = s; smax[t] = mx;
  __syncthreads();
  for (int off = 128; off > 0; off >>= 1) {
    if (t < off) { ssum[t] += ssum[t + off]; smax[t] = fmaxf(smax[t], smax[t + off]); }
    __syncthreads();
  }
  if (t == 0) { avgp[bc] = ssum[0] * (1.f / 4096.f); mxp[bc] = smax[0]; }
}

// ---------------------------------------------------------------------------
// Kernel 4: channel-attention MLP: sigmoid(relu(v@w1^T)@w2^T summed for avg,max)
// ---------------------------------------------------------------------------
__global__ __launch_bounds__(128)
void ca_kernel(const float* __restrict__ avgp, const float* __restrict__ mxp,
               const float* __restrict__ w1, const float* __restrict__ w2,
               float* __restrict__ ca) {
  __shared__ float av[128], mv[128], ha[8], hm[8];
  int b = blockIdx.x, c = threadIdx.x;
  av[c] = avgp[b * 128 + c];
  mv[c] = mxp[b * 128 + c];
  __syncthreads();
  if (c < 8) {
    float s = 0.f;
    for (int i = 0; i < 128; ++i) s += av[i] * w1[c * 128 + i];
    ha[c] = fmaxf(s, 0.f);
  } else if (c < 16) {
    int h = c - 8; float s = 0.f;
    for (int i = 0; i < 128; ++i) s += mv[i] * w1[h * 128 + i];
    hm[h] = fmaxf(s, 0.f);
  }
  __syncthreads();
  float s = 0.f;
  #pragma unroll
  for (int h = 0; h < 8; ++h) s += (ha[h] + hm[h]) * w2[c * 8 + h];
  ca[b * 128 + c] = 1.f / (1.f + __expf(-s));
}

// ---------------------------------------------------------------------------
// Kernel 5: spatial stats over channels (stride 4096) of F*ca: mean and max.
// ---------------------------------------------------------------------------
__global__ __launch_bounds__(256)
void sstat_kernel(const float* __restrict__ F, const float* __restrict__ ca,
                  float* __restrict__ s0, float* __restrict__ s1) {
  int idx = blockIdx.x * 256 + threadIdx.x;        // b*4096 + l4
  int b = idx >> 12, l4 = idx & 4095;
  const float* base = F + (size_t)b * 524288 + l4;
  const float* cab  = ca + b * 128;
  float s = 0.f, mx = -3.402823466e38f;
  for (int c4 = 0; c4 < 128; ++c4) {
    float v = base[(size_t)c4 * 4096] * cab[c4];
    s += v; mx = fmaxf(mx, v);
  }
  s0[idx] = s * (1.f / 128.f);
  s1[idx] = mx;
}

// ---------------------------------------------------------------------------
// Kernel 6: 7-tap conv (only kh=3 row survives padding) + sigmoid -> siga.
// ---------------------------------------------------------------------------
__global__ __launch_bounds__(256)
void sa_kernel(const float* __restrict__ s0, const float* __restrict__ s1,
               const float* __restrict__ convw, float* __restrict__ siga) {
  int idx = blockIdx.x * 256 + threadIdx.x;        // b*4096 + l4
  int b = idx >> 12, l4 = idx & 4095;
  float acc = 0.f;
  #pragma unroll
  for (int j = 0; j < 7; ++j) {
    int p = l4 + j - 3;
    if ((unsigned)p < 4096u) {
      int q = (b << 12) + p;
      acc += convw[21 + j] * s0[q] + convw[70 + j] * s1[q];  // [0][ci][3][j]
    }
  }
  siga[idx] = 1.f / (1.f + __expf(-acc));
}

// ---------------------------------------------------------------------------
// Kernel 7: fused epilogue: out = x + F * ca[c4] * siga[l4]  (raw reshape)
// ---------------------------------------------------------------------------
__global__ __launch_bounds__(256)
void final_kernel(const float* __restrict__ x, const float* __restrict__ F,
                  const float* __restrict__ ca, const float* __restrict__ siga,
                  float* __restrict__ out) {
  size_t idx = (size_t)blockIdx.x * 256 + threadIdx.x;   // b*524288 + f
  int b  = (int)(idx >> 19);
  int f  = (int)(idx & 524287);
  int c4 = f >> 12;
  int l4 = f & 4095;
  out[idx] = x[idx] + F[idx] * ca[b * 128 + c4] * siga[(b << 12) + l4];
}

// ---------------------------------------------------------------------------
extern "C" void kernel_launch(void* const* d_in, const int* in_sizes, int n_in,
                              void* d_out, int out_size, void* d_ws, size_t ws_size,
                              hipStream_t stream) {
  const float* x     = (const float*)d_in[0];
  const float* n1g   = (const float*)d_in[1];
  const float* n1b   = (const float*)d_in[2];
  const float* fk1c  = (const float*)d_in[3];
  const float* fk1b  = (const float*)d_in[4];
  const float* n2g   = (const float*)d_in[5];
  const float* n2b   = (const float*)d_in[6];
  const float* fk2c  = (const float*)d_in[7];
  const float* fk2b  = (const float*)d_in[8];
  const float* w1    = (const float*)d_in[9];
  const float* w2    = (const float*)d_in[10];
  const float* convw = (const float*)d_in[11];
  float* out = (float*)d_out;

  // workspace layout (~33.8 MB)
  char* ws = (char*)d_ws;
  _Float16* Bpack = (_Float16*)ws;                         // 1 MB
  float* F    = (float*)(ws + (1u << 20));                 // 32 MB
  float* avgp = (float*)(ws + (1u << 20) + (32u << 20));   // 8 KB
  float* mxp  = avgp + 2048;
  float* ca   = mxp + 2048;
  float* s0   = ca + 2048;                                 // 256 KB
  float* s1   = s0 + 65536;                                // 256 KB
  float* siga = s1 + 65536;                                // 256 KB

  pack_coeffs_kernel<<<2048, 256, 0, stream>>>(fk1c, fk2c, Bpack);
  fkan_main_kernel<<<TILES / 4, 128, 0, stream>>>(x, n1g, n1b, fk1b,
                                                  n2g, n2b, fk2b, Bpack, F);
  pool_kernel<<<16 * 128, 256, 0, stream>>>(F, avgp, mxp);
  ca_kernel<<<16, 128, 0, stream>>>(avgp, mxp, w1, w2, ca);
  sstat_kernel<<<(16 * 4096) / 256, 256, 0, stream>>>(F, ca, s0, s1);
  sa_kernel<<<(16 * 4096) / 256, 256, 0, stream>>>(s0, s1, convw, siga);
  final_kernel<<<(16 * 524288) / 256, 256, 0, stream>>>(x, F, ca, siga, out);
}
